// SelfAttention_37022618091957
// MI455X (gfx1250) — compile-verified, matrix operand order checked by reading, not run on hardware
//
#include <hip/hip_runtime.h>

#define HDIM 1024
#define SEQ  2048
#define NB   4

typedef __attribute__((ext_vector_type(16))) _Float16 v16h;
typedef __attribute__((ext_vector_type(8)))  float    v8f;
typedef unsigned int u32;
typedef __attribute__((ext_vector_type(4))) u32 u32x4;

union FragU { v16h h; u32x4 q[2]; };

// Load a 16x32 f16 WMMA operand fragment from a K-major (row-major, ld = row
// stride in elements) matrix. Works for A (rows = M) and for B when the
// source is B^T stored row-major (rows = N), per the CDNA5 16-bit layouts:
//   lane L: row = L&15, half = L>>4
//   VGPR v in 0..3  -> K = half*8  + 2v, 2v+1   (one contiguous b128)
//   VGPR v in 4..7  -> K = 16+half*8 + ...      (one contiguous b128)
__device__ __forceinline__ v16h load_frag_k(const _Float16* __restrict__ p,
                                            int ld, int lane) {
  const int r    = lane & 15;
  const int half = (lane >> 4) & 1;
  const _Float16* base = p + (size_t)r * ld + half * 8;
  FragU u;
  u.q[0] = *reinterpret_cast<const u32x4*>(base);
  u.q[1] = *reinterpret_cast<const u32x4*>(base + 16);
  return u.h;
}

__device__ __forceinline__ v8f wmma_f16(v16h a, v16h b, v8f c) {
  return __builtin_amdgcn_wmma_f32_16x16x32_f16(false, a, false, b,
                                                (short)0, c, false, false);
}

// ---------------- f32 -> f16 convert pass (x and the three weights) --------
__global__ void __launch_bounds__(256) cvt_kernel(const float* __restrict__ x,
                                                  const float* __restrict__ wq,
                                                  const float* __restrict__ wk,
                                                  const float* __restrict__ wv,
                                                  _Float16* __restrict__ xh,
                                                  _Float16* __restrict__ wh) {
  const size_t NX = (size_t)NB * SEQ * HDIM;
  const size_t NW = (size_t)HDIM * HDIM;
  size_t i = (size_t)blockIdx.x * blockDim.x + threadIdx.x;
  if (i < NX) { xh[i] = (_Float16)x[i]; return; }
  size_t j = i - NX;
  if (j < NW) { wh[j] = (_Float16)wq[j]; return; }
  j -= NW;
  if (j < NW) { wh[NW + j] = (_Float16)wk[j]; return; }
  j -= NW;
  if (j < NW) { wh[2 * NW + j] = (_Float16)wv[j]; }
}

// ---------------- Q/K/V projections ----------------------------------------
// 2 row-tiles x 3 weight matrices per wave: 5 fragment loads feed 6 WMMAs.
// Q[r,d] = sum_h x[r,h] W[d,h] + b[d].  Stores Q,K row-major f16; V transposed
// to Vt[b][d][s] so the PV GEMM gets a K-major B operand.
__global__ void __launch_bounds__(128) qkv_kernel(
    const _Float16* __restrict__ xh, const _Float16* __restrict__ wh,
    const float* __restrict__ bq, const float* __restrict__ bk,
    const float* __restrict__ bv,
    _Float16* __restrict__ Qh, _Float16* __restrict__ Kh,
    _Float16* __restrict__ Vt) {
  const int lane = threadIdx.x & 31;
  const int tile = blockIdx.x * 4 + (threadIdx.x >> 5);
  const int row0 = (tile >> 6) * 32;   // over NB*SEQ rows, 2 sub-tiles
  const int col0 = (tile & 63) * 16;   // over HDIM cols

  const _Float16* wqr = wh + (size_t)col0 * HDIM;
  const _Float16* wkr = wqr + (size_t)HDIM * HDIM;
  const _Float16* wvr = wkr + (size_t)HDIM * HDIM;
  const _Float16* xr0 = xh + (size_t)row0 * HDIM;
  const _Float16* xr1 = xr0 + (size_t)16 * HDIM;

  v8f acc[2][3] = {};
  for (int k0 = 0; k0 < HDIM; k0 += 32) {
    v16h a0 = load_frag_k(xr0 + k0, HDIM, lane);
    v16h a1 = load_frag_k(xr1 + k0, HDIM, lane);
    v16h fq = load_frag_k(wqr + k0, HDIM, lane);
    v16h fk = load_frag_k(wkr + k0, HDIM, lane);
    v16h fv = load_frag_k(wvr + k0, HDIM, lane);
    acc[0][0] = wmma_f16(a0, fq, acc[0][0]);
    acc[0][1] = wmma_f16(a0, fk, acc[0][1]);
    acc[0][2] = wmma_f16(a0, fv, acc[0][2]);
    acc[1][0] = wmma_f16(a1, fq, acc[1][0]);
    acc[1][1] = wmma_f16(a1, fk, acc[1][1]);
    acc[1][2] = wmma_f16(a1, fv, acc[1][2]);
  }

  const int n  = lane & 15;
  const int mo = (lane >> 4) * 8;
  const float vbq = bq[col0 + n], vbk = bk[col0 + n], vbv = bv[col0 + n];
  const int b  = row0 >> 11;            // row0 / SEQ  (32 | SEQ, no crossing)
  const int s0 = row0 & (SEQ - 1);
#pragma unroll
  for (int i = 0; i < 2; ++i) {
#pragma unroll
    for (int v = 0; v < 8; ++v) {
      const int m = 16 * i + v + mo;
      const size_t r = (size_t)(row0 + m);
      Qh[r * HDIM + col0 + n] = (_Float16)(acc[i][0][v] + vbq);
      Kh[r * HDIM + col0 + n] = (_Float16)(acc[i][1][v] + vbk);
      Vt[(size_t)b * HDIM * SEQ + (size_t)(col0 + n) * SEQ + (s0 + m)] =
          (_Float16)(acc[i][2][v] + vbv);
    }
  }
}

// ---------------- scores = (Q K^T)/sqrt(H), 4x2 register blocking ----------
// 64x32 output per wave: 6 fragment loads feed 8 WMMAs.
__global__ void __launch_bounds__(128) scores_kernel(
    const _Float16* __restrict__ Qh, const _Float16* __restrict__ Kh,
    _Float16* __restrict__ P) {
  const int lane = threadIdx.x & 31;
  const int tile = blockIdx.x * 4 + (threadIdx.x >> 5); // b*32*64 + qt*64 + kt
  const int b    = tile >> 11;
  const int qt   = (tile >> 6) & 31;
  const int kt   = tile & 63;
  const int row0 = qt * 64, col0 = kt * 32;

  const _Float16* qp[4];
#pragma unroll
  for (int i = 0; i < 4; ++i)
    qp[i] = Qh + ((size_t)b * SEQ + row0 + 16 * i) * HDIM;
  const _Float16* kp0 = Kh + ((size_t)b * SEQ + col0) * HDIM;
  const _Float16* kp1 = kp0 + (size_t)16 * HDIM;

  v8f acc[4][2] = {};
  for (int h0 = 0; h0 < HDIM; h0 += 32) {
    __builtin_prefetch(qp[0] + h0 + 128, 0, 0);
    v16h b0 = load_frag_k(kp0 + h0, HDIM, lane);
    v16h b1 = load_frag_k(kp1 + h0, HDIM, lane);
#pragma unroll
    for (int i = 0; i < 4; ++i) {
      v16h a = load_frag_k(qp[i] + h0, HDIM, lane);
      acc[i][0] = wmma_f16(a, b0, acc[i][0]);
      acc[i][1] = wmma_f16(a, b1, acc[i][1]);
    }
  }

  const int n  = lane & 15;
  const int mo = (lane >> 4) * 8;
  _Float16* out = P + (size_t)b * SEQ * SEQ;
  const float sc = 0.03125f;  // 1/sqrt(1024)
#pragma unroll
  for (int i = 0; i < 4; ++i) {
#pragma unroll
    for (int v = 0; v < 8; ++v) {
      const int m = row0 + 16 * i + v + mo;
      out[(size_t)m * SEQ + col0 + n]      = (_Float16)(acc[i][0][v] * sc);
      out[(size_t)m * SEQ + col0 + 16 + n] = (_Float16)(acc[i][1][v] * sc);
    }
  }
}

// ---------------- in-place row softmax over 2048 keys ----------------------
__global__ void __launch_bounds__(256) softmax_kernel(_Float16* __restrict__ P) {
  __shared__ float red[256];
  _Float16* base = P + (size_t)blockIdx.x * SEQ;
  const int t = threadIdx.x;
  float vals[8];
  float m = -1e30f;
#pragma unroll
  for (int i = 0; i < 8; ++i) {
    vals[i] = (float)base[t + i * 256];
    m = fmaxf(m, vals[i]);
  }
  red[t] = m; __syncthreads();
  for (int off = 128; off > 0; off >>= 1) {
    if (t < off) red[t] = fmaxf(red[t], red[t + off]);
    __syncthreads();
  }
  const float rowmax = red[0];
  __syncthreads();
  float s = 0.f;
#pragma unroll
  for (int i = 0; i < 8; ++i) { vals[i] = __expf(vals[i] - rowmax); s += vals[i]; }
  red[t] = s; __syncthreads();
  for (int off = 128; off > 0; off >>= 1) {
    if (t < off) red[t] += red[t + off];
    __syncthreads();
  }
  const float inv = 1.0f / red[0];
#pragma unroll
  for (int i = 0; i < 8; ++i) base[t + i * 256] = (_Float16)(vals[i] * inv);
}

// ---------------- out = P @ V, 4x2 register blocking, f32 store ------------
__global__ void __launch_bounds__(128) pv_kernel(
    const _Float16* __restrict__ P, const _Float16* __restrict__ Vt,
    float* __restrict__ out) {
  const int lane = threadIdx.x & 31;
  const int tile = blockIdx.x * 4 + (threadIdx.x >> 5); // b*32*32 + qt*32 + dt
  const int b    = tile >> 10;
  const int qt   = (tile >> 5) & 31;
  const int dt   = tile & 31;
  const int row0 = qt * 64, d0 = dt * 32;

  const _Float16* pp[4];
#pragma unroll
  for (int i = 0; i < 4; ++i)
    pp[i] = P + (size_t)b * SEQ * SEQ + (size_t)(row0 + 16 * i) * SEQ;
  const _Float16* vp0 = Vt + (size_t)b * HDIM * SEQ + (size_t)d0 * SEQ;
  const _Float16* vp1 = vp0 + (size_t)16 * SEQ;

  v8f acc[4][2] = {};
  for (int k0 = 0; k0 < SEQ; k0 += 32) {
    __builtin_prefetch(pp[0] + k0 + 128, 0, 0);
    v16h b0 = load_frag_k(vp0 + k0, SEQ, lane);
    v16h b1 = load_frag_k(vp1 + k0, SEQ, lane);
#pragma unroll
    for (int i = 0; i < 4; ++i) {
      v16h a = load_frag_k(pp[i] + k0, SEQ, lane);
      acc[i][0] = wmma_f16(a, b0, acc[i][0]);
      acc[i][1] = wmma_f16(a, b1, acc[i][1]);
    }
  }

  const int n  = lane & 15;
  const int mo = (lane >> 4) * 8;
#pragma unroll
  for (int i = 0; i < 4; ++i) {
#pragma unroll
    for (int v = 0; v < 8; ++v) {
      const int m = row0 + 16 * i + v + mo;
      out[((size_t)b * SEQ + m) * HDIM + d0 + n]      = acc[i][0][v];
      out[((size_t)b * SEQ + m) * HDIM + d0 + 16 + n] = acc[i][1][v];
    }
  }
}

extern "C" void kernel_launch(void* const* d_in, const int* in_sizes, int n_in,
                              void* d_out, int out_size, void* d_ws,
                              size_t ws_size, hipStream_t stream) {
  const float* x  = (const float*)d_in[0];
  const float* Wq = (const float*)d_in[1];
  const float* bq = (const float*)d_in[2];
  const float* Wk = (const float*)d_in[3];
  const float* bk = (const float*)d_in[4];
  const float* Wv = (const float*)d_in[5];
  const float* bv = (const float*)d_in[6];
  float* out = (float*)d_out;

  const size_t NX = (size_t)NB * SEQ * HDIM;   // 8,388,608
  const size_t NW = (size_t)HDIM * HDIM;       // 1,048,576
  _Float16* xh = (_Float16*)d_ws;
  _Float16* wh = xh + NX;
  _Float16* Qh = wh + 3 * NW;
  _Float16* Kh = Qh + NX;
  _Float16* Vt = Kh + NX;
  _Float16* P  = Vt + NX;   // NB*SEQ*SEQ f16 probs/scores (in-place softmax)

  {
    const size_t NT = NX + 3 * NW;
    cvt_kernel<<<dim3((unsigned)((NT + 255) / 256)), 256, 0, stream>>>(
        x, Wq, Wk, Wv, xh, wh);
  }
  // (NB*SEQ/32)*(HDIM/16) = 16384 waves / 4 per block
  qkv_kernel<<<dim3(4096), 128, 0, stream>>>(xh, wh, bq, bk, bv, Qh, Kh, Vt);
  // NB*(SEQ/64)*(SEQ/32) = 8192 waves / 4 per block
  scores_kernel<<<dim3(2048), 128, 0, stream>>>(Qh, Kh, P);
  softmax_kernel<<<dim3(NB * SEQ), 256, 0, stream>>>(P);
  // NB*(SEQ/64)*(HDIM/32) = 4096 waves / 4 per block
  pv_kernel<<<dim3(1024), 128, 0, stream>>>(P, Vt, out);
}
